// HNetMixerModel_28312424415326
// MI455X (gfx1250) — matrix operand order repaired and checked by reading, not compiled, not run on hardware
//
#include <hip/hip_runtime.h>
#include <cstddef>

// ---------------- model dims ----------------
#define B_  2
#define L_  1024
#define D_  1024
#define H_  16
#define NL_ 2
#define DH_ 64
#define EPS_ 1e-5f

#define USE_TDM 1   // stage GEMM tiles with the Tensor Data Mover

typedef __bf16 bf16_t;
typedef __attribute__((ext_vector_type(16))) bf16_t v16bf;
typedef __attribute__((ext_vector_type(8)))  float  v8f;
typedef __attribute__((ext_vector_type(4)))  unsigned int u32x4;
typedef __attribute__((ext_vector_type(8)))  int    i32x8;
typedef __attribute__((ext_vector_type(4)))  int    i32x4;

// ---------------- workspace layout (in floats) ----------------
#define SZ_ACT   ((size_t)B_ * L_ * D_)          // 2,097,152
#define OFF_X    ((size_t)0)
#define OFF_H    (OFF_X   + SZ_ACT)              // compressed stream / residual
#define OFF_HN   (OFF_H   + SZ_ACT)              // rmsnorm output
#define OFF_AT   (OFF_HN  + SZ_ACT)              // attention output (B,L,H,DH)
#define OFF_QUP  (OFF_AT  + SZ_ACT)              // upsampler q
#define OFF_KV   (OFF_QUP + SZ_ACT)              // upsampler kv (B,L,2D)
#define OFF_QKV  (OFF_KV  + 2 * SZ_ACT)          // layer qkv (B,L,3D)
#define OFF_RQ   (OFF_QKV)                       // routing q (aliases qkv; consumed first)
#define OFF_RK   (OFF_QKV + SZ_ACT)              // routing k
#define OFF_MISC (OFF_QKV + 3 * SZ_ACT)          // mask/dest ints + kvbias floats

// =================================================================
// 1) embedding gather
// =================================================================
__global__ void embed_kernel(const int* __restrict__ ids,
                             const float* __restrict__ emb,
                             float* __restrict__ x) {
  int bl = blockIdx.x;
  const float* e = emb + (size_t)ids[bl] * D_;
  float* xr = x + (size_t)bl * D_;
  for (int d = threadIdx.x; d < D_; d += blockDim.x) xr[d] = e[d];
}

// =================================================================
// TDM: DMA a 64-row x 32-dword fp32 tile (row stride K dwords) into LDS,
// hardware-padded +2 dwords every 32 -> LDS row stride 34 dwords.
// 2D tensor -> descriptor groups 2/3 are zero. Wave-level op.
// =================================================================
#if USE_TDM
__device__ __forceinline__ void tdm_load_tile(const float* gsrc,
                                              unsigned lds_byte_off, int K) {
  const unsigned long long ga = (unsigned long long)(const void*)gsrc;
  u32x4 g0;
  g0[0] = 1u;                                        // count=1 (valid user D#)
  g0[1] = lds_byte_off;                              // lds_addr
  g0[2] = (unsigned)(ga & 0xffffffffull);            // global_addr[31:0]
  g0[3] = (unsigned)((ga >> 32) & 0x01ffffffull)     // global_addr[56:32]
          | 0x80000000u;                             // type=2 ("image")
  i32x8 g1;
  g1[0] = (int)((2u << 16)      // data_size = 4B
              | (1u << 20)      // pad_enable
              | (4u << 22)      // pad_interval: every 32 dwords
              | (1u << 25));    // pad_amount: 2 dwords
  g1[1] = (int)(((unsigned)K & 0xffffu) << 16);      // tensor_dim0 lo16
  g1[2] = (int)((((unsigned)K >> 16) & 0xffffu)      // tensor_dim0 hi16
              | (0xffffu << 16));                    // tensor_dim1 lo16 (large)
  g1[3] = (int)(32u << 16);                          // tensor_dim1 hi | tile_dim0=32
  g1[4] = 64;                                        // tile_dim1=64, tile_dim2=0
  g1[5] = K;                                         // tensor_dim0_stride lo32
  g1[6] = 0;
  g1[7] = 0;
  const i32x4 z4 = {0, 0, 0, 0};
  const i32x8 z8 = {0, 0, 0, 0, 0, 0, 0, 0};
  __builtin_amdgcn_tensor_load_to_lds(g0, g1, z4, z4, z8, 0);
}
#endif

// =================================================================
// 2) generic WMMA GEMM:  C[M,N] = A[M,K] @ W[N,K]^T (+bias) (+resid)
//    block = 256 threads (8 waves), tile 64x64, K-step 32, bf16 WMMA
// =================================================================
__global__ __launch_bounds__(256) void gemm_kernel(
    const float* __restrict__ A, const float* __restrict__ W,
    const float* __restrict__ bias, const float* __restrict__ resid,
    float* __restrict__ C, int M, int N, int K) {
  __shared__ float As[64][34];
  __shared__ float Ws[64][34];
  const int tid  = threadIdx.x;
  const int lane = tid & 31;
  const int wave = tid >> 5;      // 0..7
  const int waveM = wave >> 2;    // 0..1  (32 rows each)
  const int waveN = wave & 3;     // 0..3  (16 cols each)
  const int bm = blockIdx.y * 64;
  const int bn = blockIdx.x * 64;
  const int l16 = lane & 15;
  const int half = lane >> 4;

  v8f acc0 = {}; v8f acc1 = {};

#if !USE_TDM
  const int row = tid >> 2;            // 0..63
  const int c4  = (tid & 3) * 8;       // 0,8,16,24
#endif

  for (int k0 = 0; k0 < K; k0 += 32) {
#if USE_TDM
    if (tid < 32) {   // wave 0 drives the Tensor Data Mover
      tdm_load_tile(A + (size_t)bm * K + k0, (unsigned)(size_t)&As[0][0], K);
      tdm_load_tile(W + (size_t)bn * K + k0, (unsigned)(size_t)&Ws[0][0], K);
      __builtin_prefetch(A + (size_t)bm * K + k0 + 32, 0, 1);  // next k-tile
      __builtin_prefetch(W + (size_t)bn * K + k0 + 32, 0, 1);
      __builtin_amdgcn_s_wait_tensorcnt(0);
    }
    __syncthreads();
#else
    {
      const float* srcA = A + (size_t)(bm + row) * K + k0 + c4;
      const float* srcW = W + (size_t)(bn + row) * K + k0 + c4;
#pragma unroll
      for (int j = 0; j < 8; ++j) As[row][c4 + j] = srcA[j];
#pragma unroll
      for (int j = 0; j < 8; ++j) Ws[row][c4 + j] = srcW[j];
    }
    __syncthreads();
#endif

    // B fragment: lane holds column n = waveN*16 + l16; K = ((e&8)<<1) + half*8 + (e&7)
    v16bf bfrag;
    {
      const int n = waveN * 16 + l16;
#pragma unroll
      for (int e = 0; e < 16; ++e) bfrag[e] = (bf16_t)Ws[n][((e & 8) << 1) + (half << 3) + (e & 7)];
    }
#pragma unroll
    for (int sub = 0; sub < 2; ++sub) {
      v16bf afrag;
      const int m = waveM * 32 + sub * 16 + l16;
#pragma unroll
      for (int e = 0; e < 16; ++e) {
        const int kk = ((e & 8) << 1) + (half << 3) + (((e >> 1) & 3) << 1) + (e & 1);
        afrag[e] = (bf16_t)As[m][kk];
      }
      if (sub == 0)
        acc0 = __builtin_amdgcn_wmma_f32_16x16x32_bf16(false, afrag, false, bfrag,
                                                       (short)0, acc0, false, false);
      else
        acc1 = __builtin_amdgcn_wmma_f32_16x16x32_bf16(false, afrag, false, bfrag,
                                                       (short)0, acc1, false, false);
    }
    __syncthreads();
  }

  const int ncol = bn + waveN * 16 + l16;
  const float bv = bias ? bias[ncol] : 0.0f;
  const int mrow0 = bm + waveM * 32 + (half << 3);
  if (resid) {
#pragma unroll
    for (int sub = 0; sub < 2; ++sub) {
      const v8f acc = sub ? acc1 : acc0;
      const int mrow = mrow0 + sub * 16;
#pragma unroll
      for (int r = 0; r < 8; ++r) {
        const size_t idx = (size_t)(mrow + r) * N + ncol;
        C[idx] = acc[r] + bv + resid[idx];
      }
    }
  } else {
#pragma unroll
    for (int sub = 0; sub < 2; ++sub) {
      const v8f acc = sub ? acc1 : acc0;
      const int mrow = mrow0 + sub * 16;
#pragma unroll
      for (int r = 0; r < 8; ++r) {
        const size_t idx = (size_t)(mrow + r) * N + ncol;
        C[idx] = acc[r] + bv;
      }
    }
  }
}

// =================================================================
// 3) routing boundary probability (one wave per position)
// =================================================================
__global__ __launch_bounds__(256) void route_kernel(const float* __restrict__ Q,
                                                    const float* __restrict__ Kp,
                                                    int* __restrict__ mask) {
  const int idx = blockIdx.x * 8 + (threadIdx.x >> 5);   // b*L + l
  const int lane = threadIdx.x & 31;
  const int l = idx % L_;
  int m = 1;
  if (l != 0) {
    const float* q = Q + (size_t)idx * D_;
    const float* k = Kp + (size_t)(idx - 1) * D_;
    float dot = 0.f, nq = 0.f, nk = 0.f;
    for (int d = lane; d < D_; d += 32) {
      const float a = q[d], c = k[d];
      dot += a * c; nq += a * a; nk += c * c;
    }
#pragma unroll
    for (int off = 16; off >= 1; off >>= 1) {
      dot += __shfl_xor(dot, off, 32);
      nq  += __shfl_xor(nq,  off, 32);
      nk  += __shfl_xor(nk,  off, 32);
    }
    const float nrm = fmaxf(sqrtf(nq) * sqrtf(nk), 1.1920929e-07f);
    const float p = 0.5f * (1.0f - dot / nrm);
    m = (rintf(p) > 0.5f) ? 1 : 0;   // straight-through round
  }
  if (lane == 0) mask[idx] = m;
}

// =================================================================
// 4) per-batch scan -> dest slots + kv validity bias
// =================================================================
__global__ __launch_bounds__(1024) void scan_kernel(const int* __restrict__ mask,
                                                    int* __restrict__ dest,
                                                    float* __restrict__ kvbias) {
  __shared__ int sc[L_];
  const int b = blockIdx.x, t = threadIdx.x;
  const int m = mask[b * L_ + t];
  sc[t] = m;
  __syncthreads();
  for (int off = 1; off < L_; off <<= 1) {
    const int v = (t >= off) ? sc[t - off] : 0;
    __syncthreads();
    sc[t] += v;
    __syncthreads();
  }
  const int total = sc[L_ - 1];
  dest[b * L_ + t] = m ? (sc[t] - 1) : -1;
  kvbias[b * L_ + t] = (t < total) ? 0.0f : -1e9f;
}

__global__ void zero_kernel(float* __restrict__ p, size_t n) {
  const size_t i = (size_t)blockIdx.x * blockDim.x + threadIdx.x;
  if (i < n) p[i] = 0.0f;
}

__global__ void scatter_kernel(const float* __restrict__ x,
                               const int* __restrict__ dest,
                               float* __restrict__ comp) {
  const int idx = blockIdx.x;
  const int d = dest[idx];
  if (d < 0) return;
  const int b = idx / L_;
  const float* src = x + (size_t)idx * D_;
  float* dst = comp + ((size_t)(b * L_ + d)) * D_;
  for (int j = threadIdx.x; j < D_; j += blockDim.x) dst[j] = src[j];
}

// =================================================================
// 5) row RMSNorm over D
// =================================================================
__global__ __launch_bounds__(256) void rms_kernel(const float* __restrict__ in,
                                                  const float* __restrict__ w,
                                                  float* __restrict__ out) {
  __shared__ float red[8];
  const int row = blockIdx.x;
  const float* r = in + (size_t)row * D_;
  float s = 0.f;
  for (int d = threadIdx.x; d < D_; d += 256) { const float v = r[d]; s += v * v; }
#pragma unroll
  for (int off = 16; off >= 1; off >>= 1) s += __shfl_xor(s, off, 32);
  if ((threadIdx.x & 31) == 0) red[threadIdx.x >> 5] = s;
  __syncthreads();
  if (threadIdx.x < 8) {
    float t = red[threadIdx.x];
#pragma unroll
    for (int off = 4; off >= 1; off >>= 1) t += __shfl_xor(t, off, 32);
    if (threadIdx.x == 0) red[0] = t;
  }
  __syncthreads();
  const float inv = rsqrtf(red[0] / (float)D_ + EPS_);
  float* o = out + (size_t)row * D_;
  for (int d = threadIdx.x; d < D_; d += 256) o[d] = r[d] * inv * w[d];
}

// =================================================================
// 6) per-head RMSNorm (+ optional RoPE); one wave per (b,l,h)
// =================================================================
__global__ __launch_bounds__(32) void headnorm_kernel(float* __restrict__ buf,
                                                      const float* __restrict__ w,
                                                      int row_stride, int base_off,
                                                      int apply_rope) {
  const int idx = blockIdx.x;               // b*L*H + l*H + h
  const int h = idx % H_;
  const int bl = idx / H_;
  const int l = bl % L_;
  const int lane = threadIdx.x;
  float* p = buf + (size_t)bl * row_stride + base_off + h * DH_;
  float a = p[lane], c = p[lane + 32];
  float ss = a * a + c * c;
#pragma unroll
  for (int off = 16; off >= 1; off >>= 1) ss += __shfl_xor(ss, off, 32);
  const float inv = rsqrtf(ss / (float)DH_ + EPS_);
  a *= inv * w[lane];
  c *= inv * w[lane + 32];
  if (apply_rope) {
    // inv_freq = 10000^(-2*lane/DH);  er[d] = l * inv_freq[d % 32]
    const float invf = __expf(-((float)(2 * lane) / (float)DH_) * 9.210340371976184f);
    const float t = (float)l * invf;
    const float ct = __cosf(t), st = __sinf(t);
    const float na = a * ct - c * st;   // q*cos + rot_half(q)*sin (low half)
    const float nc = c * ct + a * st;   // high half
    a = na; c = nc;
  }
  p[lane] = a;
  p[lane + 32] = c;
}

// =================================================================
// 7) flash attention with WMMA (masked over kv validity, no causal)
//    grid: (L/64, H, B); block: 128 (4 waves, 16 queries each)
// =================================================================
__global__ __launch_bounds__(128) void attn_kernel(
    const float* __restrict__ Qb, const float* __restrict__ Kb,
    const float* __restrict__ Vb, const float* __restrict__ kvbias,
    float* __restrict__ Ob, int q_stride, int k_stride, int v_stride) {
  __shared__ bf16_t Ks[32][DH_ + 2];
  __shared__ bf16_t Vs[32][DH_ + 2];
  __shared__ bf16_t Ps[4][16][34];

  const int b = blockIdx.z, h = blockIdx.y;
  const int tid = threadIdx.x, lane = tid & 31, wave = tid >> 5;
  const int l16 = lane & 15, half = lane >> 4;
  const int qbase = blockIdx.x * 64 + wave * 16;
  const float scale = 0.125f;   // 1/sqrt(DH)

  // Q fragments (two 32-wide k-steps over DH=64), scale folded in
  v16bf aq[2];
  {
    const int m = qbase + l16;
    const float* qrow = Qb + (size_t)(b * L_ + m) * q_stride + h * DH_;
#pragma unroll
    for (int s = 0; s < 2; ++s)
#pragma unroll
      for (int e = 0; e < 16; ++e) {
        const int kk = s * 32 + ((e & 8) << 1) + (half << 3) + (((e >> 1) & 3) << 1) + (e & 1);
        aq[s][e] = (bf16_t)(qrow[kk] * scale);
      }
  }

  v8f o0 = {}, o1 = {}, o2 = {}, o3 = {};
  float Mr[8], Lr[8];
#pragma unroll
  for (int r = 0; r < 8; ++r) { Mr[r] = -3.0e38f; Lr[r] = 0.0f; }

  for (int kc = 0; kc < L_; kc += 32) {
    __syncthreads();
    {
      const int key = tid >> 2;           // 0..31
      const int d0 = (tid & 3) * 16;
      const float* krow = Kb + (size_t)(b * L_ + kc + key) * k_stride + h * DH_ + d0;
      const float* vrow = Vb + (size_t)(b * L_ + kc + key) * v_stride + h * DH_ + d0;
#pragma unroll
      for (int j = 0; j < 16; ++j) {
        Ks[key][d0 + j] = (bf16_t)krow[j];
        Vs[key][d0 + j] = (bf16_t)vrow[j];
      }
    }
    __syncthreads();

    // S = Q @ K^T  (two 16-key column tiles)
    v8f s0 = {}, s1 = {};
#pragma unroll
    for (int s = 0; s < 2; ++s) {
      v16bf bk0, bk1;
#pragma unroll
      for (int e = 0; e < 16; ++e) {
        const int dd = s * 32 + ((e & 8) << 1) + (half << 3) + (e & 7);
        bk0[e] = Ks[l16][dd];
        bk1[e] = Ks[16 + l16][dd];
      }
      s0 = __builtin_amdgcn_wmma_f32_16x16x32_bf16(false, aq[s], false, bk0, (short)0, s0, false, false);
      s1 = __builtin_amdgcn_wmma_f32_16x16x32_bf16(false, aq[s], false, bk1, (short)0, s1, false, false);
    }

    const float bias0 = kvbias[b * L_ + kc + l16];
    const float bias1 = kvbias[b * L_ + kc + 16 + l16];
#pragma unroll
    for (int r = 0; r < 8; ++r) { s0[r] += bias0; s1[r] += bias1; }

    // online softmax per row (row = r + 8*half; reductions stay in 16-lane half)
#pragma unroll
    for (int r = 0; r < 8; ++r) {
      float mx = fmaxf(s0[r], s1[r]);
#pragma unroll
      for (int off = 8; off >= 1; off >>= 1) mx = fmaxf(mx, __shfl_xor(mx, off, 32));
      const float mnew = fmaxf(Mr[r], mx);
      const float p0 = __expf(s0[r] - mnew);
      const float p1 = __expf(s1[r] - mnew);
      float rs = p0 + p1;
#pragma unroll
      for (int off = 8; off >= 1; off >>= 1) rs += __shfl_xor(rs, off, 32);
      const float alpha = __expf(Mr[r] - mnew);
      Lr[r] = Lr[r] * alpha + rs;
      Mr[r] = mnew;
      o0[r] *= alpha; o1[r] *= alpha; o2[r] *= alpha; o3[r] *= alpha;
      const int prow = r + (half << 3);
      Ps[wave][prow][l16]      = (bf16_t)p0;
      Ps[wave][prow][16 + l16] = (bf16_t)p1;
    }

    // O += P @ V  (K-dim = 32 keys, four 16-wide d tiles)
    v16bf pa;
#pragma unroll
    for (int e = 0; e < 16; ++e) {
      const int kk = ((e & 8) << 1) + (half << 3) + (((e >> 1) & 3) << 1) + (e & 1);
      pa[e] = Ps[wave][l16][kk];
    }
#pragma unroll
    for (int t = 0; t < 4; ++t) {
      v16bf bv;
#pragma unroll
      for (int e = 0; e < 16; ++e) bv[e] = Vs[((e & 8) << 1) + (half << 3) + (e & 7)][t * 16 + l16];
      if (t == 0)      o0 = __builtin_amdgcn_wmma_f32_16x16x32_bf16(false, pa, false, bv, (short)0, o0, false, false);
      else if (t == 1) o1 = __builtin_amdgcn_wmma_f32_16x16x32_bf16(false, pa, false, bv, (short)0, o1, false, false);
      else if (t == 2) o2 = __builtin_amdgcn_wmma_f32_16x16x32_bf16(false, pa, false, bv, (short)0, o2, false, false);
      else             o3 = __builtin_amdgcn_wmma_f32_16x16x32_bf16(false, pa, false, bv, (short)0, o3, false, false);
    }
  }

  // normalize and store to (B, L, H*DH)
#pragma unroll
  for (int r = 0; r < 8; ++r) {
    const float inv = 1.0f / Lr[r];
    const int m = qbase + r + (half << 3);
    float* orow = Ob + (size_t)(b * L_ + m) * D_ + h * DH_;
    orow[0 * 16 + l16] = o0[r] * inv;
    orow[1 * 16 + l16] = o1[r] * inv;
    orow[2 * 16 + l16] = o2[r] * inv;
    orow[3 * 16 + l16] = o3[r] * inv;
  }
}

// =================================================================
// host-side orchestration
// =================================================================
extern "C" void kernel_launch(void* const* d_in, const int* in_sizes, int n_in,
                              void* d_out, int out_size, void* d_ws, size_t ws_size,
                              hipStream_t stream) {
  const int*   input_ids = (const int*)  d_in[0];
  const float* emb       = (const float*)d_in[1];
  const float* rout_wq   = (const float*)d_in[2];
  const float* rout_wk   = (const float*)d_in[3];
  const float* norm_w    = (const float*)d_in[4];
  const float* qkv_w     = (const float*)d_in[5];
  const float* qkv_b     = (const float*)d_in[6];
  const float* out_w     = (const float*)d_in[7];
  const float* out_b     = (const float*)d_in[8];
  const float* qn_w      = (const float*)d_in[9];
  const float* kn_w      = (const float*)d_in[10];
  const float* up_norm_w = (const float*)d_in[11];
  const float* up_q_w    = (const float*)d_in[12];
  const float* up_q_b    = (const float*)d_in[13];
  const float* up_kv_w   = (const float*)d_in[14];
  const float* up_kv_b   = (const float*)d_in[15];
  const float* up_out_w  = (const float*)d_in[16];
  const float* up_out_b  = (const float*)d_in[17];
  const float* up_qn_w   = (const float*)d_in[18];
  const float* up_kn_w   = (const float*)d_in[19];
  float* out = (float*)d_out;

  float* ws   = (float*)d_ws;
  float* x    = ws + OFF_X;
  float* hbuf = ws + OFF_H;     // compressed stream (residual)
  float* hn   = ws + OFF_HN;
  float* atb  = ws + OFF_AT;
  float* qup  = ws + OFF_QUP;
  float* kvb  = ws + OFF_KV;
  float* qkv  = ws + OFF_QKV;
  float* rq   = ws + OFF_RQ;    // aliases qkv region (used before it)
  float* rk   = ws + OFF_RK;
  int*   mask   = (int*)(ws + OFF_MISC);
  int*   dest   = mask + B_ * L_;
  float* kvbias = (float*)(dest + B_ * L_);

  const int M = B_ * L_;   // 2048

  // 1) embedding
  embed_kernel<<<M, 256, 0, stream>>>(input_ids, emb, x);

  // 2) routing projections (bf16 WMMA GEMMs, TDM-staged tiles)
  gemm_kernel<<<dim3(D_ / 64, M / 64), 256, 0, stream>>>(x, rout_wq, nullptr, nullptr, rq, M, D_, D_);
  gemm_kernel<<<dim3(D_ / 64, M / 64), 256, 0, stream>>>(x, rout_wk, nullptr, nullptr, rk, M, D_, D_);

  // 3) boundary decision + 4) scan + compaction
  route_kernel<<<M / 8, 256, 0, stream>>>(rq, rk, mask);
  scan_kernel<<<B_, 1024, 0, stream>>>(mask, dest, kvbias);
  zero_kernel<<<(int)((SZ_ACT + 255) / 256), 256, 0, stream>>>(hbuf, SZ_ACT);
  scatter_kernel<<<M, 256, 0, stream>>>(x, dest, hbuf);

  // 5) transformer layers on compressed stream
  for (int i = 0; i < NL_; ++i) {
    rms_kernel<<<M, 256, 0, stream>>>(hbuf, norm_w + (size_t)i * D_, hn);
    gemm_kernel<<<dim3(3 * D_ / 64, M / 64), 256, 0, stream>>>(
        hn, qkv_w + (size_t)i * 3 * D_ * D_, qkv_b + (size_t)i * 3 * D_, nullptr, qkv, M, 3 * D_, D_);
    headnorm_kernel<<<M * H_, 32, 0, stream>>>(qkv, qn_w + (size_t)i * DH_, 3 * D_, 0,  1);
    headnorm_kernel<<<M * H_, 32, 0, stream>>>(qkv, kn_w + (size_t)i * DH_, 3 * D_, D_, 1);
    attn_kernel<<<dim3(L_ / 64, H_, B_), 128, 0, stream>>>(
        qkv, qkv + D_, qkv + 2 * D_, kvbias, atb, 3 * D_, 3 * D_, 3 * D_);
    gemm_kernel<<<dim3(D_ / 64, M / 64), 256, 0, stream>>>(
        atb, out_w + (size_t)i * D_ * D_, out_b + (size_t)i * D_, hbuf, hbuf, M, D_, D_);
  }

  // 6) cross-attention upsampler
  rms_kernel<<<M, 256, 0, stream>>>(x, up_norm_w, hn);
  gemm_kernel<<<dim3(D_ / 64, M / 64), 256, 0, stream>>>(hn, up_q_w, up_q_b, nullptr, qup, M, D_, D_);
  headnorm_kernel<<<M * H_, 32, 0, stream>>>(qup, up_qn_w, D_, 0, 0);
  gemm_kernel<<<dim3(2 * D_ / 64, M / 64), 256, 0, stream>>>(hbuf, up_kv_w, up_kv_b, nullptr, kvb, M, 2 * D_, D_);
  headnorm_kernel<<<M * H_, 32, 0, stream>>>(kvb, up_kn_w, 2 * D_, 0, 0);
  attn_kernel<<<dim3(L_ / 64, H_, B_), 128, 0, stream>>>(
      qup, kvb, kvb + D_, kvbias, atb, D_, 2 * D_, 2 * D_);
  gemm_kernel<<<dim3(D_ / 64, M / 64), 256, 0, stream>>>(atb, up_out_w, up_out_b, x, out, M, D_, D_);

  (void)in_sizes; (void)n_in; (void)out_size; (void)ws_size;
}
